// SinkhornEuclideanLoss_88029649699216
// MI455X (gfx1250) — matrix operand order repaired and checked by reading, not compiled, not run on hardware
//
#include <hip/hip_runtime.h>
#include <hip/hip_bf16.h>

#define NPTS 320
#define MS   (NPTS * NPTS)
#define EPSK 0.1f

typedef __attribute__((ext_vector_type(2)))  float  v2f;
typedef __attribute__((ext_vector_type(8)))  float  v8f;
typedef __attribute__((ext_vector_type(8)))  __bf16 v8bf;
typedef __attribute__((ext_vector_type(16))) __bf16 v16bf;

// ---------------------------------------------------------------------------
// Gram + cdist fused: D[i,j] = max(sqrt(max(sq[i]+sq[j]-2*(X X^T)[i,j],0)),1e-8)
// f32 WMMA 16x16x4, one wave per 16x16 tile (20x20 tiles, 400 waves).
// ---------------------------------------------------------------------------
__global__ void gram_dist_kernel(const float* __restrict__ X, const float* __restrict__ sq,
                                 float* __restrict__ Dm, int Kdim) {
    int wave = (int)((blockIdx.x * blockDim.x + threadIdx.x) >> 5);
    int lane = (int)(threadIdx.x & 31);
    int tm = wave / 20, tn = wave - tm * 20;
    int half = lane >> 4, idx = lane & 15;

    const float* arow = X + (size_t)(tm * 16 + idx) * (size_t)Kdim;
    const float* brow = X + (size_t)(tn * 16 + idx) * (size_t)Kdim;

    v8f acc = {};
    for (int k = 0; k < Kdim; k += 4) {
        int kk = k + 2 * half;                    // lanes 0-15: K=k,k+1 ; 16-31: K=k+2,k+3
        v2f af = *(const v2f*)(arow + kk);
        v2f bf = *(const v2f*)(brow + kk);
        acc = __builtin_amdgcn_wmma_f32_16x16x4_f32(false, af, false, bf,
                                                    (short)0, acc, false, false);
    }

    int n0 = tn * 16 + idx;
    int m0 = tm * 16 + 8 * half;
    float sqn = sq[n0];
#pragma unroll
    for (int r = 0; r < 8; ++r) {
        int m = m0 + r;
        float d2 = sq[m] + sqn - 2.f * acc[r];
        Dm[(size_t)m * NPTS + n0] = fmaxf(sqrtf(fmaxf(d2, 0.f)), 1e-8f);
    }
}

// ---------------------------------------------------------------------------
// Fused Sinkhorn half-step (split-bf16 GEMM + log/exp epilogue).
//   M = U @ K   (K symmetric, both stored as hi/lo bf16 pairs)
//   v = eps*logP_sel - eps*log(M) ;  fg_out = v ; U_out = split_bf16(exp(v))
// Error-compensated product: Ah*Bh + Ah*Bl + Al*Bh with f32 accumulation.
// 60x20 tiles (960x320 batched state), one wave per tile.
// ---------------------------------------------------------------------------
__global__ void sink_step_kernel(const __bf16* __restrict__ Uhi, const __bf16* __restrict__ Ulo,
                                 const __bf16* __restrict__ Khi, const __bf16* __restrict__ Klo,
                                 const float* __restrict__ lPl, const float* __restrict__ lPo,
                                 int isG,
                                 float* __restrict__ fg_out,
                                 __bf16* __restrict__ UhiO, __bf16* __restrict__ UloO) {
    int wave = (int)((blockIdx.x * blockDim.x + threadIdx.x) >> 5);
    int lane = (int)(threadIdx.x & 31);
    int tm = wave / 20, tn = wave - tm * 20;
    int half = lane >> 4, idx = lane & 15;

    const __bf16* arh = Uhi + (size_t)(tm * 16 + idx) * NPTS;
    const __bf16* arl = Ulo + (size_t)(tm * 16 + idx) * NPTS;
    const __bf16* brh = Khi + (size_t)(tn * 16 + idx) * NPTS;
    const __bf16* brl = Klo + (size_t)(tn * 16 + idx) * NPTS;

    v8f acc = {};
    for (int k = 0; k < NPTS; k += 32) {
        // A-frag: lanes 0-15 hold K=k..k+7 (v0-3) and k+16..k+23 (v4-7); lanes 16-31 +8
        int ka0 = k + 8 * half, ka1 = k + 16 + 8 * half;
        v8bf ah0 = *(const v8bf*)(arh + ka0);
        v8bf ah1 = *(const v8bf*)(arh + ka1);
        v8bf al0 = *(const v8bf*)(arl + ka0);
        v8bf al1 = *(const v8bf*)(arl + ka1);
        v16bf ah = __builtin_shufflevector(ah0, ah1, 0,1,2,3,4,5,6,7,8,9,10,11,12,13,14,15);
        v16bf al = __builtin_shufflevector(al0, al1, 0,1,2,3,4,5,6,7,8,9,10,11,12,13,14,15);
        // B-frag: same K order as A-frag (row-major sym K)
        v8bf bh0 = *(const v8bf*)(brh + ka0);
        v8bf bh1 = *(const v8bf*)(brh + ka1);
        v8bf bl0 = *(const v8bf*)(brl + ka0);
        v8bf bl1 = *(const v8bf*)(brl + ka1);
        v16bf bh = __builtin_shufflevector(bh0, bh1, 0,1,2,3,4,5,6,7,8,9,10,11,12,13,14,15);
        v16bf bl = __builtin_shufflevector(bl0, bl1, 0,1,2,3,4,5,6,7,8,9,10,11,12,13,14,15);
        acc = __builtin_amdgcn_wmma_f32_16x16x32_bf16(false, ah, false, bh, (short)0, acc, false, false);
        acc = __builtin_amdgcn_wmma_f32_16x16x32_bf16(false, ah, false, bl, (short)0, acc, false, false);
        acc = __builtin_amdgcn_wmma_f32_16x16x32_bf16(false, al, false, bh, (short)0, acc, false, false);
    }

    int n0 = tn * 16 + idx;
    int m0 = tm * 16 + 8 * half;
    int t  = tm / 20;                       // batch id (0..2), uniform per tile
    const float* lsel = isG ? ((t < 2) ? lPl : lPo)    // log_b per batch
                            : ((t == 1) ? lPl : lPo);  // log_a per batch
#pragma unroll
    for (int r = 0; r < 8; ++r) {
        int m = m0 + r;                                // global row [0,960)
        size_t o = (size_t)m * NPTS + n0;
        size_t w = (size_t)(m - t * NPTS) * NPTS + n0; // row within batch
        float val = EPSK * lsel[w] - EPSK * logf(fmaxf(acc[r], 1e-30f));
        fg_out[o] = val;
        float u = expf(val);
        __bf16 h = (__bf16)u;
        UhiO[o] = h;
        UloO[o] = (__bf16)(u - (float)h);
    }
}

// ---------------------------------------------------------------------------
__global__ void rowsq_kernel(const float* __restrict__ X, int D, float* __restrict__ out) {
    __shared__ float s[128];
    int row = blockIdx.x;
    float acc = 0.f;
    for (int j = threadIdx.x; j < D; j += 128) { float v = X[(size_t)row * D + j]; acc += v * v; }
    s[threadIdx.x] = acc; __syncthreads();
    for (int off = 64; off > 0; off >>= 1) {
        if ((int)threadIdx.x < off) s[threadIdx.x] += s[threadIdx.x + off];
        __syncthreads();
    }
    if (threadIdx.x == 0) out[row] = s[0];
}

__global__ void max_kernel(const float* __restrict__ Dm, float* __restrict__ scal) {
    __shared__ float s[1024];
    float m = 0.f;
    for (int e = threadIdx.x; e < MS; e += 1024) m = fmaxf(m, Dm[e]);
    s[threadIdx.x] = m; __syncthreads();
    for (int off = 512; off > 0; off >>= 1) {
        if ((int)threadIdx.x < off) s[threadIdx.x] = fmaxf(s[threadIdx.x], s[threadIdx.x + off]);
        __syncthreads();
    }
    if (threadIdx.x == 0) scal[0] = s[0];
}

// Lower median (rank 51199 0-based) via value bisection over [0, maxD].
__global__ void median_kernel(const float* __restrict__ Dm, float* __restrict__ scal) {
    __shared__ int s[1024];
    float lo = 0.f, hi = scal[0];
    for (int it = 0; it < 64; ++it) {
        float mid = 0.5f * (lo + hi);
        int c = 0;
        for (int e = threadIdx.x; e < MS; e += 1024) c += (Dm[e] <= mid) ? 1 : 0;
        s[threadIdx.x] = c; __syncthreads();
        for (int off = 512; off > 0; off >>= 1) {
            if ((int)threadIdx.x < off) s[threadIdx.x] += s[threadIdx.x + off];
            __syncthreads();
        }
        int total = s[0]; __syncthreads();
        if (total >= (MS / 2)) hi = mid; else lo = mid;
    }
    if (threadIdx.x == 0) {
        float med = hi;
        scal[1] = (med == 0.f) ? 1e-6f : med;
    }
}

// K = exp(-C/eps) split to hi/lo bf16; S_orig,S_lat (diag zeroed) into Po/Pl.
__global__ void sim_kernel(const float* __restrict__ Dor, const float* __restrict__ Dla,
                           const float* __restrict__ scal, const float* __restrict__ log_sigma_lat,
                           __bf16* __restrict__ Khi, __bf16* __restrict__ Klo,
                           float* __restrict__ So, float* __restrict__ Sl) {
    int e = blockIdx.x * 256 + threadIdx.x;
    int i = e / NPTS, j = e - i * NPTS;
    float maxd  = fmaxf(scal[0], 1e-8f);
    float sig_o = scal[1];
    float ls    = log_sigma_lat[0];
    float sp    = (ls > 20.f) ? ls : log1pf(expf(ls));   // softplus
    float sig_l = sp + 1e-6f;
    float dor = Dor[e], dla = Dla[e];
    float kv = expf(-(dor / maxd) / EPSK);
    __bf16 kh = (__bf16)kv;
    Khi[e] = kh;
    Klo[e] = (__bf16)(kv - (float)kh);
    float so = expf(-dor * dor / (2.f * sig_o * sig_o));
    float sl = expf(-dla * dla / (2.f * sig_l * sig_l));
    if (i == j) { so = 0.f; sl = 0.f; }
    So[e] = so; Sl[e] = sl;
}

// Row-normalize P in place; logP = log(clip(P,1e-8)).
__global__ void rownorm_kernel(float* __restrict__ P, float* __restrict__ logP) {
    __shared__ float s[128];
    int row = blockIdx.x;
    float* pr = P + (size_t)row * NPTS;
    float acc = 0.f;
    for (int j = threadIdx.x; j < NPTS; j += 128) acc += pr[j];
    s[threadIdx.x] = acc; __syncthreads();
    for (int off = 64; off > 0; off >>= 1) {
        if ((int)threadIdx.x < off) s[threadIdx.x] += s[threadIdx.x + off];
        __syncthreads();
    }
    float inv = 1.f / fmaxf(s[0], 1e-8f);
    for (int j = threadIdx.x; j < NPTS; j += 128) {
        float p = pr[j] * inv;
        pr[j] = p;
        logP[(size_t)row * NPTS + j] = logf(fmaxf(p, 1e-8f));
    }
}

// U0 = exp(f0=0) = 1 (hi=1, lo=0)
__global__ void fillu_kernel(__bf16* __restrict__ Uhi, __bf16* __restrict__ Ulo) {
    int e = blockIdx.x * 256 + threadIdx.x;
    Uhi[e] = (__bf16)1.0f;
    Ulo[e] = (__bf16)0.0f;
}

// cost_rows[t*320+i] = <a_row, f_row> + <b_row, g_row>
__global__ void cost_kernel(const float* __restrict__ fb, const float* __restrict__ gb,
                            const float* __restrict__ Po, const float* __restrict__ Pl,
                            float* __restrict__ costR) {
    __shared__ float s[128];
    int bi = blockIdx.x;
    int t = bi / NPTS, i = bi - t * NPTS;
    const float* ar = ((t == 1) ? Pl : Po) + (size_t)i * NPTS;
    const float* br = ((t < 2)  ? Pl : Po) + (size_t)i * NPTS;
    const float* fr = fb + (size_t)bi * NPTS;
    const float* gr = gb + (size_t)bi * NPTS;
    float acc = 0.f;
    for (int k = threadIdx.x; k < NPTS; k += 128)
        acc += ar[k] * fr[k] + br[k] * gr[k];
    s[threadIdx.x] = acc; __syncthreads();
    for (int off = 64; off > 0; off >>= 1) {
        if ((int)threadIdx.x < off) s[threadIdx.x] += s[threadIdx.x + off];
        __syncthreads();
    }
    if (threadIdx.x == 0) costR[bi] = s[0];
}

__global__ void recon_kernel(const float* __restrict__ xo, const float* __restrict__ xr,
                             float* __restrict__ part) {
    __shared__ float s[1024];
    int e = blockIdx.x * 1024 + threadIdx.x;
    float d = xr[e] - xo[e];
    s[threadIdx.x] = d * d; __syncthreads();
    for (int off = 512; off > 0; off >>= 1) {
        if ((int)threadIdx.x < off) s[threadIdx.x] += s[threadIdx.x + off];
        __syncthreads();
    }
    if (threadIdx.x == 0) part[blockIdx.x] = s[0];
}

__global__ void final_kernel(const float* __restrict__ costR, const float* __restrict__ part,
                             const float* __restrict__ lvr, const float* __restrict__ lvt,
                             float* __restrict__ out) {
    float ot[3];
    for (int t = 0; t < 3; ++t) {
        float a = 0.f;
        for (int i = 0; i < NPTS; ++i) a += costR[t * NPTS + i];
        ot[t] = a / (float)NPTS;
    }
    float topo = fmaxf(ot[0] - 0.5f * ot[1] - 0.5f * ot[2], 0.f) * 1.0f;  // TOPO_MULT
    float rs = 0.f;
    for (int i = 0; i < 640; ++i) rs += part[i];
    float recon = rs / (320.f * 2048.f);
    float pr = expf(-lvr[0]);
    float pt = expf(-lvt[0]);
    out[0] = 0.5f * pr * recon + 0.5f * lvr[0] + 0.5f * pt * topo + 0.5f * lvt[0];
    out[1] = recon;
    out[2] = topo;
}

// ---------------------------------------------------------------------------
extern "C" void kernel_launch(void* const* d_in, const int* in_sizes, int n_in,
                              void* d_out, int out_size, void* d_ws, size_t ws_size,
                              hipStream_t stream) {
    const float* xo  = (const float*)d_in[0];   // 320 x 2048
    const float* xr  = (const float*)d_in[1];   // 320 x 2048
    const float* zl  = (const float*)d_in[2];   // 320 x 128
    const float* lsl = (const float*)d_in[3];   // log_sigma_lat
    const float* lvr = (const float*)d_in[4];   // log_var_recon
    const float* lvt = (const float*)d_in[5];   // log_var_topo
    float* out = (float*)d_out;                 // [total, recon, topo]

    float* W = (float*)d_ws;
    float* sqO   = W;               // 320
    float* sqL   = W + 320;         // 320
    float* scal  = W + 640;         // [0]=maxD, [1]=median sigma
    float* part  = W + 704;         // 640 recon partials
    float* costR = W + 1344;        // 960
    float* base  = W + 2304;
    float* Dor = base + 0 * MS;
    float* Dla = base + 1 * MS;
    float* Po  = base + 2 * MS;
    float* Pl  = base + 3 * MS;
    float* lPo = base + 4 * MS;
    float* lPl = base + 5 * MS;
    float* fb  = base + 6 * MS;     // 3*MS
    float* gb  = base + 9 * MS;     // 3*MS
    __bf16* B16  = (__bf16*)(base + 12 * MS);
    __bf16* Khi  = B16;             // MS
    __bf16* Klo  = B16 + 1 * MS;    // MS
    __bf16* UhiA = B16 + 2 * MS;    // 3*MS
    __bf16* UloA = B16 + 5 * MS;    // 3*MS
    __bf16* UhiB = B16 + 8 * MS;    // 3*MS
    __bf16* UloB = B16 + 11 * MS;   // 3*MS  (total ~7.8 MB)

    // Pairwise distances (f32 WMMA Gram with fused cdist epilogue)
    rowsq_kernel<<<NPTS, 128, 0, stream>>>(xo, 2048, sqO);
    rowsq_kernel<<<NPTS, 128, 0, stream>>>(zl, 128, sqL);
    gram_dist_kernel<<<50, 256, 0, stream>>>(xo, sqO, Dor, 2048);
    gram_dist_kernel<<<50, 256, 0, stream>>>(zl, sqL, Dla, 128);

    // max, lower-median, Gibbs kernels + split-bf16 K, row-stochastic P / logP
    max_kernel<<<1, 1024, 0, stream>>>(Dor, scal);
    median_kernel<<<1, 1024, 0, stream>>>(Dor, scal);
    sim_kernel<<<400, 256, 0, stream>>>(Dor, Dla, scal, lsl, Khi, Klo, Po, Pl);
    rownorm_kernel<<<NPTS, 128, 0, stream>>>(Po, lPo);
    rownorm_kernel<<<NPTS, 128, 0, stream>>>(Pl, lPl);

    // Batched Sinkhorn: 3 problems stacked (960x320), split-bf16 WMMA + fused update.
    fillu_kernel<<<1200, 256, 0, stream>>>(UhiA, UloA);
    for (int it = 0; it < 50; ++it) {
        sink_step_kernel<<<150, 256, 0, stream>>>(UhiA, UloA, Khi, Klo, lPl, lPo, 1,
                                                  gb, UhiB, UloB);   // g-update
        sink_step_kernel<<<150, 256, 0, stream>>>(UhiB, UloB, Khi, Klo, lPl, lPo, 0,
                                                  fb, UhiA, UloA);   // f-update
    }

    // Losses
    recon_kernel<<<640, 1024, 0, stream>>>(xo, xr, part);
    cost_kernel<<<960, 128, 0, stream>>>(fb, gb, Po, Pl, costR);
    final_kernel<<<1, 1, 0, stream>>>(costR, part, lvr, lvt, out);
}
